// ScaledDotProductAttention_83434034692674
// MI455X (gfx1250) — compile-verified, matrix operand order checked by reading, not run on hardware
//
#include <hip/hip_runtime.h>

// ---------------------------------------------------------------------------
// MI455X (gfx1250) per-point KNN attention.
// Bandwidth-bound (AI ~0.5 flop/byte, ~1.06 GB traffic, floor ~45us @23.3TB/s).
// One wave32 = one point. Both GEMVs run on the WMMA pipe
// (v_wmma_f32_16x16x32_f16) with the replicated-operand trick; softmax stays
// in exact f32 VALU. K/V are streamed once with non-temporal loads.
//
// Layout assumptions (from CDNA5 ISA 7.12.2):
//  A 16-bit 16x32: lane-half h picks K-indices kidx(e) = (e<8 ? e : e+8) + 8h.
//  B 16-bit 32x16 (mirroring the documented SWMMAC B table): lane = column n,
//    lane-half h and vector element e pick contraction row kidx(e).
//  D f32 16x16: VGPR0 = row (M=0 | M=8), N = lane&15  -> with replicated A all
//    rows are equal, so element 0 of the accumulator is the wanted value.
// ---------------------------------------------------------------------------

typedef __attribute__((ext_vector_type(16))) _Float16 v16h;
typedef __attribute__((ext_vector_type(8)))  float    v8f;
typedef __attribute__((ext_vector_type(4)))  float    f32x4;

#define DK 128
#define KN 32
#define DV 128
#define WAVES_PER_BLOCK 8

__device__ __forceinline__ f32x4 nt_load4(const float* p) {
  return __builtin_nontemporal_load((const f32x4*)p);
}
__device__ __forceinline__ float nt_load1(const float* p) {
  return __builtin_nontemporal_load(p);
}

// pack four f32x4 runs (two contiguous 8-float runs) into a v16h so that
// element e maps to run0[e] for e<8 and run1[e-8] for e>=8.
__device__ __forceinline__ v16h pack16(f32x4 u0, f32x4 u1, f32x4 u2, f32x4 u3) {
  v16h r;
#pragma unroll
  for (int e = 0; e < 4; ++e) {
    r[e]      = (_Float16)u0[e];
    r[4 + e]  = (_Float16)u1[e];
    r[8 + e]  = (_Float16)u2[e];
    r[12 + e] = (_Float16)u3[e];
  }
  return r;
}

__global__ __launch_bounds__(256)
void knn_attn_kernel(const float* __restrict__ Q, const float* __restrict__ K,
                     const float* __restrict__ V, float* __restrict__ ctxOut,
                     float* __restrict__ attnOut, int P) {
  __shared__ float qsmem[WAVES_PER_BLOCK][DK];  // per-wave q broadcast buffer
  __shared__ float asmem[WAVES_PER_BLOCK][KN];  // per-wave attn broadcast buffer

  const int lane = threadIdx.x & 31;
  const int w    = threadIdx.x >> 5;
  const int n    = lane & 15;   // WMMA column owned by this lane
  const int h    = lane >> 4;   // lane-half selector for 16-bit operand layout

  const int p = blockIdx.x * WAVES_PER_BLOCK + w;
  if (p >= P) return;

  const float* Qp = Q + (size_t)p * DK;
  const float* Kp = K + (size_t)p * KN * DK;
  const float* Vp = V + (size_t)p * KN * DV;
  float* qs = qsmem[w];
  float* as = asmem[w];

  // ---- stage q into LDS (coalesced b128 load, b128 LDS store) -------------
  f32x4 q4 = *(const f32x4*)(Qp + 4 * lane);
  *(f32x4*)(qs + 4 * lane) = q4;
  asm volatile("s_wait_dscnt 0" ::: "memory");  // intra-wave LDS fence

  // ---- scores s = K * q via WMMA (A = q replicated across rows) -----------
  v8f accA = {0.f, 0.f, 0.f, 0.f, 0.f, 0.f, 0.f, 0.f};  // k rows 0..15
  v8f accB = {0.f, 0.f, 0.f, 0.f, 0.f, 0.f, 0.f, 0.f};  // k rows 16..31
#pragma unroll
  for (int c = 0; c < 4; ++c) {            // 4 d-chunks of 32
    const int d0 = 32 * c;
    // A operand: q[d0 + kidx(e)] broadcast from LDS (two contiguous runs)
    v16h a = pack16(*(const f32x4*)(qs + d0 + 8 * h),
                    *(const f32x4*)(qs + d0 + 8 * h + 4),
                    *(const f32x4*)(qs + d0 + 16 + 8 * h),
                    *(const f32x4*)(qs + d0 + 16 + 8 * h + 4));
    // B operand: column n = K row (n | n+16), elements at d = d0 + kidx(e).
    // Per instruction: 16 rows x 64B contiguous (both halves adjoin) -> good
    // coalescing; NT hint so 1GB stream skips L2 retention.
    const float* kr0 = Kp + (size_t)n * DK + d0 + 8 * h;
    const float* kr1 = Kp + (size_t)(n + 16) * DK + d0 + 8 * h;
    v16h b0 = pack16(nt_load4(kr0), nt_load4(kr0 + 4),
                     nt_load4(kr0 + 16), nt_load4(kr0 + 20));
    v16h b1 = pack16(nt_load4(kr1), nt_load4(kr1 + 4),
                     nt_load4(kr1 + 16), nt_load4(kr1 + 20));
    accA = __builtin_amdgcn_wmma_f32_16x16x32_f16(false, a, false, b0,
                                                  (short)0, accA, false, false);
    accB = __builtin_amdgcn_wmma_f32_16x16x32_f16(false, a, false, b1,
                                                  (short)0, accB, false, false);
  }
  // lane l holds score for neighbor l (halves pick the right accumulator)
  float s = ((h == 0) ? accA[0] : accB[0]) * 0.08838834764831845f; // 1/sqrt(128)

  // ---- softmax over the 32 neighbors (exact f32, wave32 butterflies) ------
  float mx = s;
#pragma unroll
  for (int off = 16; off > 0; off >>= 1)
    mx = fmaxf(mx, __shfl_xor(mx, off, 32));
  float ex = __expf(s - mx);
  float sm = ex;
#pragma unroll
  for (int off = 16; off > 0; off >>= 1)
    sm += __shfl_xor(sm, off, 32);
  float at = ex / sm;

  attnOut[(size_t)p * KN + lane] = at;  // coalesced 128B store
  as[lane] = at;
  asm volatile("s_wait_dscnt 0" ::: "memory");

  // ---- context = attn * V via WMMA (A = attn replicated, Kdim = 32 = KN) --
  v16h aat = pack16(*(const f32x4*)(as + 8 * h),
                    *(const f32x4*)(as + 8 * h + 4),
                    *(const f32x4*)(as + 16 + 8 * h),
                    *(const f32x4*)(as + 16 + 8 * h + 4));
  float ctxv[8];
#pragma unroll
  for (int t = 0; t < 8; ++t) {           // 8 d-tiles of 16
    const int d0 = 16 * t;
    v16h b;
#pragma unroll
    for (int e = 0; e < 8; ++e) {
      // B[k, n] = V[k, d0+n]; each b32 instruction covers two contiguous
      // 64B row segments (lanes 0-15 row 8h+e, lanes 16-31 row 8h+e+16? no:
      // halves read rows 8h+e / 16+8h+e per their own h) -> 2x64B segments.
      b[e]     = (_Float16)nt_load1(Vp + (size_t)(8 * h + e) * DV + d0 + n);
      b[8 + e] = (_Float16)nt_load1(Vp + (size_t)(16 + 8 * h + e) * DV + d0 + n);
    }
    v8f c0 = {0.f, 0.f, 0.f, 0.f, 0.f, 0.f, 0.f, 0.f};
    c0 = __builtin_amdgcn_wmma_f32_16x16x32_f16(false, aat, false, b,
                                                (short)0, c0, false, false);
    ctxv[t] = c0[0];  // = context[d0 + n] on every lane (rows replicated)
  }
  // paired store: lanes 0-15 write even tiles, 16-31 odd tiles -> 128B/instr
  float* cp = ctxOut + (size_t)p * DV;
#pragma unroll
  for (int tp = 0; tp < 4; ++tp) {
    float v = (h == 0) ? ctxv[2 * tp] : ctxv[2 * tp + 1];
    cp[32 * tp + 16 * h + n] = v;
  }
}

extern "C" void kernel_launch(void* const* d_in, const int* in_sizes, int n_in,
                              void* d_out, int out_size, void* d_ws, size_t ws_size,
                              hipStream_t stream) {
  const float* Q = (const float*)d_in[0];
  const float* K = (const float*)d_in[1];
  const float* V = (const float*)d_in[2];
  const int P = in_sizes[0] / DK;          // B*N points
  float* out     = (float*)d_out;
  float* ctxOut  = out;                    // [P, 128]
  float* attnOut = out + (size_t)P * DV;   // [P, 32] (== [B,N,1,KN] flat)

  dim3 block(256);
  dim3 grid((P + WAVES_PER_BLOCK - 1) / WAVES_PER_BLOCK);
  knn_attn_kernel<<<grid, block, 0, stream>>>(Q, K, V, ctxOut, attnOut, P);
}
